// Layer_15418932593464
// MI455X (gfx1250) — matrix operand order repaired and checked by reading, not compiled, not run on hardware
//
#include <hip/hip_runtime.h>
#include <hip/hip_bf16.h>

typedef __attribute__((ext_vector_type(16))) __bf16 v16bf;
typedef __attribute__((ext_vector_type(8)))  __bf16 v8bf;
typedef __attribute__((ext_vector_type(8)))  float  v8f;

// exact spelling from the compiler diagnostic for the async-builtin params
typedef int v4i_vs __attribute__((__vector_size__(4 * sizeof(int))));
typedef __attribute__((address_space(1))) v4i_vs as1_v4i;
typedef __attribute__((address_space(3))) v4i_vs as3_v4i;

#define DIM     1024
#define TOKENS  4096   // B*N = 2*2048
#define SEQ     2048
#define HEADS   16
#define HD      64

// ---------------------------------------------------------------- helpers

__device__ __forceinline__ __bf16 f2bf(float f) {
    unsigned u = __builtin_bit_cast(unsigned, f);
    unsigned r = (u + 0x7FFFu + ((u >> 16) & 1u)) >> 16;   // RNE
    return __builtin_bit_cast(__bf16, (unsigned short)r);
}

// Build a 16-element bf16 fragment from two 16B-aligned LDS locations.
__device__ __forceinline__ v16bf make_frag(const __bf16* p0, const __bf16* p1) {
    struct Pair { v8bf lo, hi; } s;
    s.lo = *(const v8bf*)p0;
    s.hi = *(const v8bf*)p1;
    return __builtin_bit_cast(v16bf, s);
}

// Async global -> LDS copy of 16 bytes per lane (GLOBAL_LOAD_ASYNC_TO_LDS_B128,
// tracked by ASYNCcnt).
__device__ __forceinline__ void async_copy_b128(__bf16* lds_dst, const __bf16* gsrc) {
#if __has_builtin(__builtin_amdgcn_global_load_async_to_lds_b128)
    // flat address == global address numerically (addrspace 1);
    // low 32 bits of a flat LDS pointer == LDS byte address (addrspace 3).
    as1_v4i* g = (as1_v4i*)(size_t)gsrc;
    as3_v4i* l = (as3_v4i*)(size_t)(unsigned)(size_t)lds_dst;
    __builtin_amdgcn_global_load_async_to_lds_b128(g, l, 0, 0);
#else
    unsigned loff = (unsigned)(size_t)lds_dst;
    asm volatile("global_load_async_to_lds_b128 %0, %1, off"
                 :: "v"(loff), "v"(gsrc) : "memory");
#endif
}

__device__ __forceinline__ void async_wait0() {
#if __has_builtin(__builtin_amdgcn_s_wait_asynccnt)
    __builtin_amdgcn_s_wait_asynccnt(0);
#else
    asm volatile("s_wait_asynccnt 0x0" ::: "memory");
#endif
}

// ---------------------------------------------------------------- convert

__global__ void convert_f32_to_bf16(const float* __restrict__ s,
                                    __bf16* __restrict__ d, int n4) {
    int i = blockIdx.x * blockDim.x + threadIdx.x;
    if (i >= n4) return;
    float4 v = ((const float4*)s)[i];
    __bf16* o = d + (size_t)i * 4;
    o[0] = f2bf(v.x); o[1] = f2bf(v.y); o[2] = f2bf(v.z); o[3] = f2bf(v.w);
}

// ---------------------------------------------------------------- layernorm
// one row (C=1024) per 256-thread block; writes fp32 (optional) + bf16

__global__ __launch_bounds__(256)
void layernorm_kernel(const float* __restrict__ x,
                      const float* __restrict__ w,
                      const float* __restrict__ b,
                      float* __restrict__ outf,
                      __bf16* __restrict__ outb) {
    const int row = blockIdx.x;
    const int tid = threadIdx.x;
    const float* xr = x + (size_t)row * DIM;

    float4 v = *(const float4*)(xr + tid * 4);
    float s1 = v.x + v.y + v.z + v.w;
    float s2 = v.x*v.x + v.y*v.y + v.z*v.z + v.w*v.w;
    #pragma unroll
    for (int off = 16; off >= 1; off >>= 1) {
        s1 += __shfl_xor(s1, off, 32);
        s2 += __shfl_xor(s2, off, 32);
    }
    __shared__ float red1[8], red2[8];
    const int wid = tid >> 5, lane = tid & 31;
    if (lane == 0) { red1[wid] = s1; red2[wid] = s2; }
    __syncthreads();
    if (tid == 0) {
        float a1 = 0.f, a2 = 0.f;
        #pragma unroll
        for (int i = 0; i < 8; ++i) { a1 += red1[i]; a2 += red2[i]; }
        red1[0] = a1; red2[0] = a2;
    }
    __syncthreads();
    const float mu   = red1[0] * (1.0f / DIM);
    const float var  = red2[0] * (1.0f / DIM) - mu * mu;
    const float rstd = rsqrtf(var + 1e-5f);

    float4 wv = *(const float4*)(w + tid * 4);
    float4 bv = *(const float4*)(b + tid * 4);
    float y0 = (v.x - mu) * rstd * wv.x + bv.x;
    float y1 = (v.y - mu) * rstd * wv.y + bv.y;
    float y2 = (v.z - mu) * rstd * wv.z + bv.z;
    float y3 = (v.w - mu) * rstd * wv.w + bv.w;

    if (outf) {
        float4 o; o.x = y0; o.y = y1; o.z = y2; o.w = y3;
        *(float4*)(outf + (size_t)row * DIM + tid * 4) = o;
    }
    __bf16* ob = outb + (size_t)row * DIM + tid * 4;
    ob[0] = f2bf(y0); ob[1] = f2bf(y1); ob[2] = f2bf(y2); ob[3] = f2bf(y3);
}

// ---------------------------------------------------------------- GEMM
// C[M,N] = A[M,K] (bf16, row-major) x W[N,K]^T (bf16, row-major torch weight)
// 256 threads = 8 waves; block tile 128x128; wave tile 32x64; K-step 32.
// Double-buffered LDS fed by async global->LDS copies (ASYNCcnt).

template<bool BIAS, bool GELU, bool RES, bool OUTF, bool OUTB>
__global__ __launch_bounds__(256)
void gemm_wmma(const __bf16* __restrict__ A, const __bf16* __restrict__ W,
               const float* __restrict__ bias, const float* __restrict__ res,
               float* __restrict__ outf, __bf16* __restrict__ outb,
               int M, int N, int K) {
    constexpr int BM = 128, BN = 128, BK = 32, LD = BK + 8;  // pad: 80B rows
    __shared__ __align__(16) __bf16 As[2][BM][LD];
    __shared__ __align__(16) __bf16 Bs[2][BN][LD];

    const int tid = threadIdx.x;
    const int bm0 = blockIdx.y * BM;
    const int bn0 = blockIdx.x * BN;
    const int lane = tid & 31, wid = tid >> 5;
    const int wm = (wid & 3) * 32;   // wave M offset
    const int wn = (wid >> 2) * 64;  // wave N offset
    const int lr = lane & 15, hf = lane >> 4;

    // tile loaders: each thread moves 16 bf16 of A and 16 of B (2x b128 each)
    const int lrow = tid >> 1;
    const int lcol = (tid & 1) * 16;
    const __bf16* Ag = A + (size_t)(bm0 + lrow) * K + lcol;
    const __bf16* Wg = W + (size_t)(bn0 + lrow) * K + lcol;

    auto load_tile = [&](int buf, int k0) {
        async_copy_b128(&As[buf][lrow][lcol],     Ag + k0);
        async_copy_b128(&As[buf][lrow][lcol + 8], Ag + k0 + 8);
        async_copy_b128(&Bs[buf][lrow][lcol],     Wg + k0);
        async_copy_b128(&Bs[buf][lrow][lcol + 8], Wg + k0 + 8);
    };

    v8f acc[2][4];
    #pragma unroll
    for (int i = 0; i < 2; ++i)
        #pragma unroll
        for (int j = 0; j < 4; ++j)
            #pragma unroll
            for (int e = 0; e < 8; ++e) acc[i][j][e] = 0.f;

    const int KT = K / BK;
    load_tile(0, 0);
    for (int kt = 0; kt < KT; ++kt) {
        async_wait0();        // our async writes for buf (kt&1) have landed
        __syncthreads();      // ... and everyone else's too
        if (kt + 1 < KT) load_tile((kt + 1) & 1, (kt + 1) * BK);
        if (kt + 2 < KT) {    // stream next-next K tile toward the caches
            __builtin_prefetch((const void*)(Ag + (kt + 2) * BK), 0, 1);
            __builtin_prefetch((const void*)(Wg + (kt + 2) * BK), 0, 1);
        }
        const int buf = kt & 1;

        v16bf af[2], bf_[4];
        #pragma unroll
        for (int tm = 0; tm < 2; ++tm) {
            const __bf16* p = &As[buf][wm + tm * 16 + lr][0];
            af[tm] = make_frag(p + 8 * hf, p + 16 + 8 * hf);     // ISA A layout
        }
        #pragma unroll
        for (int tn = 0; tn < 4; ++tn) {
            const __bf16* p = &Bs[buf][wn + tn * 16 + lr][0];
            bf_[tn] = make_frag(p + 8 * hf, p + 16 + 8 * hf);   // ISA B layout
        }
        #pragma unroll
        for (int tm = 0; tm < 2; ++tm)
            #pragma unroll
            for (int tn = 0; tn < 4; ++tn)
                acc[tm][tn] = __builtin_amdgcn_wmma_f32_16x16x32_bf16(
                    false, af[tm], false, bf_[tn], (short)0, acc[tm][tn],
                    false, false);
    }

    // epilogue: C/D layout -> row m = hf*8 + r, col n = lr
    #pragma unroll
    for (int tn = 0; tn < 4; ++tn) {
        const int gn = bn0 + wn + tn * 16 + lr;
        const float bv = BIAS ? bias[gn] : 0.f;
        #pragma unroll
        for (int tm = 0; tm < 2; ++tm) {
            #pragma unroll
            for (int r = 0; r < 8; ++r) {
                const int gm = bm0 + wm + tm * 16 + hf * 8 + r;
                float vv = acc[tm][tn][r];
                if (BIAS) vv += bv;
                if (GELU) vv = 0.5f * vv * (1.0f + erff(vv * 0.70710678118654752f));
                if (RES)  vv += res[(size_t)gm * N + gn];
                if (OUTF) outf[(size_t)gm * N + gn] = vv;
                if (OUTB) outb[(size_t)gm * N + gn] = f2bf(vv);
            }
        }
    }
}

// ---------------------------------------------------------------- attention
// flash attention: 128 threads (4 waves), 64 query rows per block,
// stream 64-row K/V tiles. S and O GEMMs via bf16 WMMA, softmax in fp32.
// Q/K staged with async global->LDS copies; V needs a transpose (manual).

__global__ __launch_bounds__(128)
void attention_kernel(const __bf16* __restrict__ qkv, __bf16* __restrict__ o) {
    constexpr int LD = HD + 8;  // 72 elements -> 144B rows (16B multiple)
    __shared__ __align__(16) __bf16 Qs[64][LD];
    __shared__ __align__(16) __bf16 Ks[64][LD];
    __shared__ __align__(16) __bf16 Vt[HD][LD];      // V transposed: [d][kv]
    __shared__ __align__(16) __bf16 Ps[4][16][LD];   // per-wave P tile

    const int tid   = threadIdx.x;
    const int q0    = blockIdx.x * 64;
    const int head  = blockIdx.y;
    const int batch = blockIdx.z;
    const int lane = tid & 31, wid = tid >> 5;
    const int lr = lane & 15, hf = lane >> 4;

    const size_t rowstride = 3 * DIM;
    const size_t base = (size_t)batch * SEQ * rowstride + (size_t)head * HD;

    const int crow = tid >> 1;          // 0..63
    const int ccol = (tid & 1) * 32;    // 0 or 32

    { // load Q block once (async; completion covered by first iteration's wait)
        const __bf16* src = qkv + base + (size_t)(q0 + crow) * rowstride + ccol;
        #pragma unroll
        for (int i = 0; i < 4; ++i)
            async_copy_b128(&Qs[crow][ccol + i * 8], src + i * 8);
    }

    float m_i[8], l_i[8];
    v8f Oacc[4];
    #pragma unroll
    for (int r = 0; r < 8; ++r) { m_i[r] = -1e30f; l_i[r] = 0.f; }
    #pragma unroll
    for (int t = 0; t < 4; ++t)
        #pragma unroll
        for (int e = 0; e < 8; ++e) Oacc[t][e] = 0.f;

    for (int j = 0; j < SEQ / 64; ++j) {
        __syncthreads();   // previous tile fully consumed
        { // stage K tile (async) and transposed V tile (manual)
            const size_t kvrow = (size_t)(j * 64 + crow) * rowstride;
            const __bf16* ksrc = qkv + base + DIM     + kvrow + ccol;
            const __bf16* vsrc = qkv + base + 2 * DIM + kvrow + ccol;
            #pragma unroll
            for (int i = 0; i < 4; ++i)
                async_copy_b128(&Ks[crow][ccol + i * 8], ksrc + i * 8);
            #pragma unroll
            for (int i = 0; i < 4; ++i) {
                v8bf vv = *(const v8bf*)(vsrc + i * 8);
                #pragma unroll
                for (int e = 0; e < 8; ++e)
                    Vt[ccol + i * 8 + e][crow] = vv[e];
            }
        }
        async_wait0();
        __syncthreads();

        // ---- S = Q * K^T (16 rows x 64 kv per wave)
        v8f s[4];
        #pragma unroll
        for (int t = 0; t < 4; ++t)
            #pragma unroll
            for (int e = 0; e < 8; ++e) s[t][e] = 0.f;
        #pragma unroll
        for (int kk = 0; kk < 2; ++kk) {
            const __bf16* qp = &Qs[wid * 16 + lr][kk * 32];
            v16bf qa = make_frag(qp + 8 * hf, qp + 16 + 8 * hf);
            #pragma unroll
            for (int tn = 0; tn < 4; ++tn) {
                const __bf16* kp = &Ks[tn * 16 + lr][kk * 32];
                v16bf kb = make_frag(kp + 8 * hf, kp + 16 + 8 * hf);
                s[tn] = __builtin_amdgcn_wmma_f32_16x16x32_bf16(
                    false, qa, false, kb, (short)0, s[tn], false, false);
            }
        }
        #pragma unroll
        for (int tn = 0; tn < 4; ++tn)
            #pragma unroll
            for (int r = 0; r < 8; ++r) s[tn][r] *= 0.125f;   // hd^-0.5

        // ---- online softmax (row m = hf*8 + r lives in one 16-lane half)
        float sc[8];
        #pragma unroll
        for (int r = 0; r < 8; ++r) {
            float mx = -1e30f;
            #pragma unroll
            for (int tn = 0; tn < 4; ++tn) mx = fmaxf(mx, s[tn][r]);
            mx = fmaxf(mx, __shfl_xor(mx, 1, 32));
            mx = fmaxf(mx, __shfl_xor(mx, 2, 32));
            mx = fmaxf(mx, __shfl_xor(mx, 4, 32));
            mx = fmaxf(mx, __shfl_xor(mx, 8, 32));
            const float mnew = fmaxf(m_i[r], mx);
            sc[r] = __expf(m_i[r] - mnew);
            m_i[r] = mnew;
            float rs = 0.f;
            #pragma unroll
            for (int tn = 0; tn < 4; ++tn) {
                float p = __expf(s[tn][r] - mnew);
                s[tn][r] = p;
                rs += p;
            }
            rs += __shfl_xor(rs, 1, 32);
            rs += __shfl_xor(rs, 2, 32);
            rs += __shfl_xor(rs, 4, 32);
            rs += __shfl_xor(rs, 8, 32);
            l_i[r] = l_i[r] * sc[r] + rs;
        }
        #pragma unroll
        for (int t = 0; t < 4; ++t)
            #pragma unroll
            for (int r = 0; r < 8; ++r) Oacc[t][r] *= sc[r];

        // ---- C-layout P -> A-layout via per-wave LDS tile
        #pragma unroll
        for (int tn = 0; tn < 4; ++tn)
            #pragma unroll
            for (int r = 0; r < 8; ++r)
                Ps[wid][hf * 8 + r][tn * 16 + lr] = f2bf(s[tn][r]);

        // ---- O += P * V
        #pragma unroll
        for (int kk = 0; kk < 2; ++kk) {
            const __bf16* pp = &Ps[wid][lr][kk * 32];
            v16bf pa = make_frag(pp + 8 * hf, pp + 16 + 8 * hf);
            #pragma unroll
            for (int td = 0; td < 4; ++td) {
                const __bf16* vp = &Vt[td * 16 + lr][kk * 32];
                v16bf vb = make_frag(vp + 8 * hf, vp + 16 + 8 * hf);
                Oacc[td] = __builtin_amdgcn_wmma_f32_16x16x32_bf16(
                    false, pa, false, vb, (short)0, Oacc[td], false, false);
            }
        }
    }

    // epilogue: O /= l, scatter to [token][head*64+d] bf16
    #pragma unroll
    for (int r = 0; r < 8; ++r) {
        const float inv = 1.0f / l_i[r];
        const int n = q0 + wid * 16 + hf * 8 + r;
        __bf16* dst = o + ((size_t)batch * SEQ + n) * DIM + head * HD;
        #pragma unroll
        for (int td = 0; td < 4; ++td)
            dst[td * 16 + lr] = f2bf(Oacc[td][r] * inv);
    }
}

// ---------------------------------------------------------------- launch

extern "C" void kernel_launch(void* const* d_in, const int* in_sizes, int n_in,
                              void* d_out, int out_size, void* d_ws, size_t ws_size,
                              hipStream_t stream) {
    (void)in_sizes; (void)n_in; (void)out_size; (void)ws_size;
    const float* x      = (const float*)d_in[0];
    const float* ln1_w  = (const float*)d_in[1];
    const float* ln1_b  = (const float*)d_in[2];
    const float* qkv_w  = (const float*)d_in[3];
    const float* proj_w = (const float*)d_in[4];
    const float* proj_b = (const float*)d_in[5];
    const float* ln2_w  = (const float*)d_in[6];
    const float* ln2_b  = (const float*)d_in[7];
    const float* fc1_w  = (const float*)d_in[8];
    const float* fc1_b  = (const float*)d_in[9];
    const float* fc2_w  = (const float*)d_in[10];
    const float* fc2_b  = (const float*)d_in[11];
    float* out = (float*)d_out;

    char* ws = (char*)d_ws;
    const size_t MB = 1024ull * 1024ull;
    float*  xn_f32   = (float*) (ws +  0 * MB);   // 16 MB  LN1 out (residual base)
    float*  x1_f32   = (float*) (ws + 16 * MB);   // 16 MB  after attn residual
    __bf16* xn_bf16  = (__bf16*)(ws + 32 * MB);   //  8 MB  LN1 bf16; reused as LN2 out
    __bf16* qkv_bf16 = (__bf16*)(ws + 40 * MB);   // 32 MB  qkv bf16; reused as fc1/gelu out
    __bf16* o_bf16   = (__bf16*)(ws + 72 * MB);   //  8 MB  attention output
    __bf16* wq_bf16  = (__bf16*)(ws + 80 * MB);   //  6 MB
    __bf16* wp_bf16  = (__bf16*)(ws + 86 * MB);   //  2 MB
    __bf16* w1_bf16  = (__bf16*)(ws + 88 * MB);   //  8 MB
    __bf16* w2_bf16  = (__bf16*)(ws + 96 * MB);   //  8 MB

    auto cvt = [&](const float* s, __bf16* d, int n) {
        int n4 = n / 4;
        convert_f32_to_bf16<<<(n4 + 255) / 256, 256, 0, stream>>>(s, d, n4);
    };
    cvt(qkv_w,  wq_bf16, 3 * DIM * DIM);
    cvt(proj_w, wp_bf16, DIM * DIM);
    cvt(fc1_w,  w1_bf16, 4096 * DIM);
    cvt(fc2_w,  w2_bf16, DIM * 4096);

    // LN1: residual base (fp32) + bf16 GEMM operand
    layernorm_kernel<<<TOKENS, 256, 0, stream>>>(x, ln1_w, ln1_b, xn_f32, xn_bf16);

    // qkv = xn @ qkv_w^T  (no bias) -> bf16
    gemm_wmma<false, false, false, false, true>
        <<<dim3(3072 / 128, TOKENS / 128), 256, 0, stream>>>(
            xn_bf16, wq_bf16, nullptr, nullptr, nullptr, qkv_bf16,
            TOKENS, 3 * DIM, DIM);

    // attention -> o_bf16
    attention_kernel<<<dim3(SEQ / 64, HEADS, 2), 128, 0, stream>>>(qkv_bf16, o_bf16);

    // x1 = xn + o @ proj_w^T + proj_b   (residual uses NORMED x per reference)
    gemm_wmma<true, false, true, true, false>
        <<<dim3(DIM / 128, TOKENS / 128), 256, 0, stream>>>(
            o_bf16, wp_bf16, proj_b, xn_f32, x1_f32, nullptr,
            TOKENS, DIM, DIM);

    // LN2 -> bf16 operand (reuse xn_bf16)
    layernorm_kernel<<<TOKENS, 256, 0, stream>>>(x1_f32, ln2_w, ln2_b, nullptr, xn_bf16);

    // h = gelu(hn @ fc1_w^T + fc1_b) -> bf16 (reuse qkv region)
    gemm_wmma<true, true, false, false, true>
        <<<dim3(4096 / 128, TOKENS / 128), 256, 0, stream>>>(
            xn_bf16, w1_bf16, fc1_b, nullptr, nullptr, qkv_bf16,
            TOKENS, 4096, DIM);

    // out = x1 + h @ fc2_w^T + fc2_b  (fp32)
    gemm_wmma<true, false, true, true, false>
        <<<dim3(DIM / 128, TOKENS / 128), 256, 0, stream>>>(
            qkv_bf16, w2_bf16, fc2_b, x1_f32, out, nullptr,
            TOKENS, DIM, 4096);
}
